// SpatialAttention_1906965480020
// MI455X (gfx1250) — matrix operand order repaired and checked by reading, not compiled, not run on hardware
//
#include <hip/hip_runtime.h>

typedef __attribute__((ext_vector_type(16))) _Float16     v16h;
typedef __attribute__((ext_vector_type(8)))  float        v8f;
typedef __attribute__((ext_vector_type(4)))  float        f32x4;
typedef __attribute__((ext_vector_type(4)))  unsigned int u32x4;

union H16 { v16h v; unsigned int u[8]; u32x4 q[2]; };

static __device__ __forceinline__ unsigned int pkh(float a, float b) {
    return __builtin_bit_cast(unsigned int, __builtin_amdgcn_cvt_pkrtz(a, b));
}

// CDNA5 async global->LDS copy (ASYNCcnt-tracked), per CDNA5 ISA.
// lds_off is the raw LDS byte offset (low 32 bits of the flat shared address).
static __device__ __forceinline__ void async_g2lds_b128(unsigned lds_off, const void* gaddr) {
    asm volatile("global_load_async_to_lds_b128 %0, %1, off"
                 :: "v"(lds_off), "v"(gaddr) : "memory");
}
static __device__ __forceinline__ void wait_async0() {
    asm volatile("s_wait_asynccnt 0" ::: "memory");
}

#define CHN 256
#define NHW 4096
#define DQK 32

// ---------------------------------------------------------------------------
// Kernel 1: fused q/k/v projections.  x (b,c,n) f32 -> q,kT (b,n,32) f16 and
// v (b,c,n) f16 via WMMA f16 GEMMs.  x tile transposed to (n,c) f16 in LDS.
// ---------------------------------------------------------------------------
__global__ __launch_bounds__(256) void proj_qkv(
    const float* __restrict__ x,
    const float* __restrict__ Wq, const float* __restrict__ bq,
    const float* __restrict__ Wk, const float* __restrict__ bk,
    const float* __restrict__ Wv, const float* __restrict__ bv,
    _Float16* __restrict__ qo, _Float16* __restrict__ ko, _Float16* __restrict__ vo)
{
    __shared__ _Float16 xT[64 * 264];   // (n, c) f16, row stride 264 halves (528B, 16B aligned)

    const int b    = blockIdx.y;
    const int n0   = blockIdx.x * 64;
    const int t    = threadIdx.x;
    const int wv   = t >> 5;
    const int lane = t & 31;
    const int lo   = lane & 15;
    const int hi   = lane >> 4;

    // ---- stage x tile transposed into LDS (f32 -> f16) ----
    const float* xb = x + (size_t)b * CHN * NHW;
    #pragma unroll
    for (int it = 0; it < 16; ++it) {
        const int c  = it * 16 + (t >> 4);
        const int nn = (t & 15) * 4;
        f32x4 d = *(const f32x4*)(xb + (size_t)c * NHW + n0 + nn);
        xT[(nn + 0) * 264 + c] = (_Float16)d.x;
        xT[(nn + 1) * 264 + c] = (_Float16)d.y;
        xT[(nn + 2) * 264 + c] = (_Float16)d.z;
        xT[(nn + 3) * 264 + c] = (_Float16)d.w;
    }
    __syncthreads();

    // 80 tile jobs: ot 0-1 -> q, 2-3 -> k (D is (n,o)); ot 4-19 -> v (D is (c,n)).
    for (int j = wv; j < 80; j += 8) {
        const int ot = j >> 2;
        const int nt = j & 3;
        if (ot < 4) {
            const float* W    = (ot < 2) ? Wq : Wk;
            const float* bias = (ot < 2) ? bq : bk;
            _Float16*    outp = (ot < 2) ? qo : ko;
            const int    o0   = (ot & 1) * 16;

            v8f C;
            const float bb = bias[o0 + lo];
            #pragma unroll
            for (int r = 0; r < 8; ++r) C[r] = bb;

            const float* Wr = W + (size_t)(o0 + lo) * CHN;
            const char*  ar = (const char*)xT + (nt * 16 + lo) * 528 + hi * 16;
            #pragma unroll
            for (int cc = 0; cc < 8; ++cc) {
                H16 A;   // A = xT rows (16 n x 32 c): A layout K {0-7,16-23}/{8-15,24-31}
                A.q[0] = *(const u32x4*)(ar + cc * 64);
                A.q[1] = *(const u32x4*)(ar + cc * 64 + 32);
                H16 Bm;  // B = W rows as columns: B layout K {0-7,16-23}/{8-15,24-31}
                const float* wp = Wr + cc * 32 + hi * 8;
                f32x4 w0 = *(const f32x4*)(wp + 0);
                f32x4 w1 = *(const f32x4*)(wp + 4);
                f32x4 w2 = *(const f32x4*)(wp + 16);
                f32x4 w3 = *(const f32x4*)(wp + 20);
                Bm.u[0] = pkh(w0.x, w0.y); Bm.u[1] = pkh(w0.z, w0.w);
                Bm.u[2] = pkh(w1.x, w1.y); Bm.u[3] = pkh(w1.z, w1.w);
                Bm.u[4] = pkh(w2.x, w2.y); Bm.u[5] = pkh(w2.z, w2.w);
                Bm.u[6] = pkh(w3.x, w3.y); Bm.u[7] = pkh(w3.z, w3.w);
                C = __builtin_amdgcn_wmma_f32_16x16x32_f16(false, A.v, false, Bm.v,
                                                           (short)0, C, false, false);
            }
            _Float16* op = outp + (size_t)b * NHW * DQK;
            #pragma unroll
            for (int r = 0; r < 8; ++r) {
                const int n = n0 + nt * 16 + r + 8 * hi;
                op[(size_t)n * DQK + o0 + lo] = (_Float16)C[r];
            }
        } else {
            const int o0 = (ot - 4) * 16;
            v8f C;
            #pragma unroll
            for (int r = 0; r < 8; ++r) C[r] = bv[o0 + r + 8 * hi];

            const float* Wr = Wv + (size_t)(o0 + lo) * CHN;
            const char*  br = (const char*)xT + (nt * 16 + lo) * 528 + hi * 16;
            #pragma unroll
            for (int cc = 0; cc < 8; ++cc) {
                H16 A;   // A = Wv rows (16 c_out x 32 c): A layout
                const float* wp = Wr + cc * 32 + hi * 8;
                f32x4 w0 = *(const f32x4*)(wp + 0);
                f32x4 w1 = *(const f32x4*)(wp + 4);
                f32x4 w2 = *(const f32x4*)(wp + 16);
                f32x4 w3 = *(const f32x4*)(wp + 20);
                A.u[0] = pkh(w0.x, w0.y); A.u[1] = pkh(w0.z, w0.w);
                A.u[2] = pkh(w1.x, w1.y); A.u[3] = pkh(w1.z, w1.w);
                A.u[4] = pkh(w2.x, w2.y); A.u[5] = pkh(w2.z, w2.w);
                A.u[6] = pkh(w3.x, w3.y); A.u[7] = pkh(w3.z, w3.w);
                H16 Bm;  // B = xT rows as columns: B layout
                Bm.q[0] = *(const u32x4*)(br + cc * 64);
                Bm.q[1] = *(const u32x4*)(br + cc * 64 + 32);
                C = __builtin_amdgcn_wmma_f32_16x16x32_f16(false, A.v, false, Bm.v,
                                                           (short)0, C, false, false);
            }
            _Float16* op = vo + (size_t)b * CHN * NHW;
            #pragma unroll
            for (int r = 0; r < 8; ++r) {
                const int c = o0 + r + 8 * hi;
                const int n = n0 + nt * 16 + lo;
                op[(size_t)c * NHW + n] = (_Float16)C[r];
            }
        }
    }
}

// ---------------------------------------------------------------------------
// Kernel 2: fused flash attention + residual epilogue.
// Per wave: 16-query tile; loop over 32-key blocks:
//   S^T = Kblk x Q  (lane = query column -> cheap softmax)
//   O^T += Vblk x P (lane = query column -> coalesced (c,n) epilogue)
// V blocks are double-buffered in LDS via async global->LDS copies (ASYNCcnt).
// ---------------------------------------------------------------------------
#define VBUF_BYTES (256 * 80)   // one V block: 256 c rows x 32 keys f16, row stride 80B

__global__ __launch_bounds__(256) void flash_attn(
    const _Float16* __restrict__ qg, const _Float16* __restrict__ kg,
    const _Float16* __restrict__ vg, const float* __restrict__ x,
    const float* __restrict__ gamma, float* __restrict__ out)
{
    __shared__ _Float16 vsh[2 * 256 * 40];  // two V-block buffers

    const int b    = blockIdx.y;
    const int nq0  = blockIdx.x * 128;
    const int t    = threadIdx.x;
    const int wv   = t >> 5;
    const int lane = t & 31;
    const int lo   = lane & 15;
    const int hi   = lane >> 4;

    const unsigned lds_base = (unsigned)(size_t)(char*)vsh;  // LDS byte offset of vsh
    const int quad = t >> 2, sub = t & 3;

    const _Float16* kbase = kg + (size_t)b * NHW * DQK;
    const _Float16* vbase = vg + (size_t)b * CHN * NHW;

    // Q tile (16 q x 32 d) in WMMA *B* layout: lane = query column, K {0-7,16-23}/{8-15,24-31}
    H16 Q;
    {
        const char* qr = (const char*)(qg + ((size_t)b * NHW + nq0 + wv * 16 + lo) * DQK) + hi * 16;
        Q.q[0] = *(const u32x4*)(qr);
        Q.q[1] = *(const u32x4*)(qr + 32);
    }

    v8f O[16];
    #pragma unroll
    for (int i = 0; i < 16; ++i)
        #pragma unroll
        for (int r = 0; r < 8; ++r) O[i][r] = 0.0f;
    float mrow = -1e30f, lrow = 0.0f;

    // ---- prologue: async-stage V block 0 into buffer 0 ----
    #pragma unroll
    for (int jj = 0; jj < 4; ++jj) {
        const int c = quad + jj * 64;
        async_g2lds_b128(lds_base + c * 80 + sub * 16,
                         (const char*)(vbase + (size_t)c * NHW) + sub * 16);
    }

    for (int kblk = 0; kblk < 128; ++kblk) {
        const int m0  = kblk * 32;
        const int cur = kblk & 1;

        // my async fills done -> signal; after barrier everyone's fills are visible
        wait_async0();
        __syncthreads();

        // ---- async-stage next V block into the other buffer ----
        if (m0 + 32 < NHW) {
            const unsigned dstb = lds_base + (1 - cur) * VBUF_BYTES;
            #pragma unroll
            for (int jj = 0; jj < 4; ++jj) {
                const int c = quad + jj * 64;
                async_g2lds_b128(dstb + c * 80 + sub * 16,
                                 (const char*)(vbase + (size_t)c * NHW + m0 + 32) + sub * 16);
            }
            // prefetch next K block (2KB, shared by all 8 waves via L2/WGP$)
            __builtin_prefetch((const char*)(kbase + (size_t)(m0 + 32 + lo) * DQK), 0, 3);
        }

        // ---- S^T tiles: A = kT rows (A layout), B = Q registers ----
        H16 K0, K1;
        {
            const char* kr = (const char*)(kbase + (size_t)(m0 + lo) * DQK) + hi * 16;
            K0.q[0] = *(const u32x4*)(kr);
            K0.q[1] = *(const u32x4*)(kr + 32);
            const char* kr1 = kr + 16 * DQK * 2;
            K1.q[0] = *(const u32x4*)(kr1);
            K1.q[1] = *(const u32x4*)(kr1 + 32);
        }
        v8f Z;
        #pragma unroll
        for (int r = 0; r < 8; ++r) Z[r] = 0.0f;
        v8f S0 = __builtin_amdgcn_wmma_f32_16x16x32_f16(false, K0.v, false, Q.v, (short)0, Z, false, false);
        v8f S1 = __builtin_amdgcn_wmma_f32_16x16x32_f16(false, K1.v, false, Q.v, (short)0, Z, false, false);

        // ---- online softmax (row = query lives in lanes {lo, lo+16}) ----
        float mx = S0[0];
        #pragma unroll
        for (int r = 1; r < 8; ++r) mx = fmaxf(mx, S0[r]);
        #pragma unroll
        for (int r = 0; r < 8; ++r) mx = fmaxf(mx, S1[r]);
        mx = fmaxf(mx, __shfl_xor(mx, 16, 32));
        const float mnew  = fmaxf(mrow, mx);
        const float alpha = __expf(mrow - mnew);
        mrow = mnew;

        float sum = 0.0f;
        unsigned int p0[4], p1[4];
        #pragma unroll
        for (int r = 0; r < 4; ++r) {
            const float a0 = __expf(S0[2 * r]     - mnew);
            const float a1 = __expf(S0[2 * r + 1] - mnew);
            const float b0 = __expf(S1[2 * r]     - mnew);
            const float b1 = __expf(S1[2 * r + 1] - mnew);
            sum += (a0 + a1) + (b0 + b1);
            p0[r] = pkh(a0, a1);
            p1[r] = pkh(b0, b1);
        }
        sum += __shfl_xor(sum, 16, 32);
        lrow = lrow * alpha + sum;

        // ---- repack P as B operand in this code's contiguous k order (same as V A loads): lanes<16 hold keys 0-15, lanes>=16 keys 16-31 ----
        H16 P;
        #pragma unroll
        for (int r = 0; r < 4; ++r) {
            const unsigned int x0 = (unsigned int)__shfl_xor((int)p0[r], 16, 32);
            const unsigned int x1 = (unsigned int)__shfl_xor((int)p1[r], 16, 32);
            P.u[r]     = hi ? x1    : p0[r];
            P.u[4 + r] = hi ? p1[r] : x0;
        }

        // ---- rescale O only when the running max actually moved (wave-uniform skip) ----
        if (__ballot(alpha < 1.0f) != 0ull) {
            #pragma unroll
            for (int ct = 0; ct < 16; ++ct)
                #pragma unroll
                for (int r = 0; r < 8; ++r) O[ct][r] *= alpha;
        }

        // ---- O^T tiles += Vblk x P ----
        const char* vbuf = (const char*)vsh + cur * VBUF_BYTES;
        #pragma unroll
        for (int ct = 0; ct < 16; ++ct) {
            H16 A;  // A = v rows (16 c x 32 k) from LDS, A layout
            const char* vr = vbuf + (ct * 16 + lo) * 80 + hi * 32;
            A.q[0] = *(const u32x4*)(vr);
            A.q[1] = *(const u32x4*)(vr + 16);
            O[ct] = __builtin_amdgcn_wmma_f32_16x16x32_f16(false, A.v, false, P.v,
                                                           (short)0, O[ct], false, false);
        }
    }

    // ---- epilogue: out[b][c][n] = x + gamma * O/l  (lane-contiguous in n) ----
    const float invl = 1.0f / lrow;
    const float g    = gamma[0];
    const int   n    = nq0 + wv * 16 + lo;
    const float* xb2 = x   + (size_t)b * CHN * NHW;
    float*       ob  = out + (size_t)b * CHN * NHW;
    #pragma unroll
    for (int ct = 0; ct < 16; ++ct) {
        #pragma unroll
        for (int r = 0; r < 8; ++r) {
            const int    c   = ct * 16 + r + 8 * hi;
            const size_t idx = (size_t)c * NHW + n;
            ob[idx] = xb2[idx] + g * (O[ct][r] * invl);
        }
    }
}

// ---------------------------------------------------------------------------
extern "C" void kernel_launch(void* const* d_in, const int* in_sizes, int n_in,
                              void* d_out, int out_size, void* d_ws, size_t ws_size,
                              hipStream_t stream) {
    (void)in_sizes; (void)n_in; (void)out_size; (void)ws_size;
    const float* x     = (const float*)d_in[0];
    const float* Wq    = (const float*)d_in[1];
    const float* bq    = (const float*)d_in[2];
    const float* Wk    = (const float*)d_in[3];
    const float* bk    = (const float*)d_in[4];
    const float* Wv    = (const float*)d_in[5];
    const float* bv    = (const float*)d_in[6];
    const float* gamma = (const float*)d_in[7];

    // workspace: q (8x4096x32) + kT (8x4096x32) + v (8x256x4096), all f16 (~20MB)
    _Float16* qws = (_Float16*)d_ws;
    _Float16* kws = qws + (size_t)8 * NHW * DQK;
    _Float16* vws = kws + (size_t)8 * NHW * DQK;

    dim3 g1(64, 8), blk(256);
    proj_qkv<<<g1, blk, 0, stream>>>(x, Wq, bq, Wk, bk, Wv, bv, qws, kws, vws);

    dim3 g2(32, 8);
    flash_attn<<<g2, blk, 0, stream>>>(qws, kws, vws, x, gamma, (float*)d_out);
}
